// CASTWithLUT_1005022347826
// MI455X (gfx1250) — compile-verified, matrix operand order checked by reading, not run on hardware
//
#include <hip/hip_runtime.h>
#include <hip/hip_bf16.h>
#include <math.h>

// ---------------- types ----------------
typedef __bf16 bf16_t;
typedef __attribute__((ext_vector_type(16))) __bf16        v16bf;
typedef __attribute__((ext_vector_type(8)))  float         v8f;
typedef __attribute__((ext_vector_type(4))) unsigned int   u32x4;
typedef __attribute__((ext_vector_type(8)))  int           i32x8;
typedef __attribute__((ext_vector_type(4)))  int           i32x4;

#define EDIM   256
#define VCODES 1024
#define NLUT   8
#define HDIM   256
#define IMGSZ  256
#define BATCH  4
#define GRID3  33
#define G3CUBE 35937           // 33^3
#define GENOUT (NLUT*3*G3CUBE) // 862488

enum { EF_SHIFT = 1, EF_SCALE = 2, EF_GELU = 4, EF_SIG = 8, EF_RES = 16 };

// ---------------- device helpers ----------------
static __device__ __forceinline__ float gelu_exact(float x) {
  return 0.5f * x * (1.f + erff(x * 0.70710678118654752f));
}

// Build a 2-D TDM descriptor (D#) per CDNA5 ISA and issue
// TENSOR_LOAD_TO_LDS. Tile = 32 (K, contiguous, bf16) x 128 rows; tensor dims are
// the *remaining* extents so TDM zero-fills out-of-bounds rows for us.
// 6-arg builtin on this toolchain:
//   (uint32x4 g0, int32x8 g1, int32x4 g2, int32x4 g3, int32x8 pad, i32 cpol)
static __device__ __forceinline__ void tdm_load_2d(unsigned ldsaddr, const bf16_t* gptr,
                                                   int rows_rem, int stride_elems) {
  unsigned long long ga = (unsigned long long)gptr;
  u32x4 g0;
  g0.x = 1u;                                            // count=1, user mode
  g0.y = ldsaddr;                                       // LDS byte address
  g0.z = (unsigned)ga;                                  // global addr [31:0]
  g0.w = ((unsigned)(ga >> 32) & 0x1FFFFFFu) | (2u << 30); // addr[56:32] | type=2
  unsigned td0 = 32u;
  unsigned td1 = rows_rem < 0 ? 0u : (unsigned)rows_rem;
  unsigned long long sb = (unsigned long long)stride_elems;
  i32x8 g1;
  g1[0] = (int)(1u << 16);                              // data_size=1 (2 bytes)
  g1[1] = (int)((td0 & 0xFFFFu) << 16);                 // tensor_dim0 lo16
  g1[2] = (int)((td0 >> 16) | ((td1 & 0xFFFFu) << 16)); // d0 hi16 | d1 lo16
  g1[3] = (int)(((td1 >> 16) & 0xFFFFu) | (32u << 16)); // d1 hi16 | tile_dim0=32
  g1[4] = (int)128u;                                    // tile_dim1=128, tile_dim2=0
  g1[5] = (int)(unsigned)(sb & 0xFFFFFFFFull);          // tensor_dim0_stride lo32
  g1[6] = (int)(unsigned)((sb >> 32) & 0xFFFFull);      // stride hi16 | dim1_stride lo=0
  g1[7] = 0;
  i32x4 zz = {0, 0, 0, 0};
  i32x8 zz8 = {0, 0, 0, 0, 0, 0, 0, 0};
  __builtin_amdgcn_tensor_load_to_lds(g0, g1, zz, zz, zz8, 0);
}

// Load a 16x32 bf16 WMMA A/B fragment from an LDS row (row-major, 32 bf16/row).
// Per ISA layout: lanes 0-15 hold K[0..7]+K[16..23], lanes 16-31 K[8..15]+K[24..31].
static __device__ __forceinline__ v16bf ld_frag(const bf16_t* p) {
  union U { v16bf v; u32x4 q[2]; };
  U u;
  u.q[0] = *(const u32x4*)(p);
  u.q[1] = *(const u32x4*)(p + 16);
  return u.v;
}

// ---------------- WMMA GEMM: C[M,N] f32 = A[M,Kp] bf16 * Bt[N,Kp]^T bf16 ----------------
// Double-buffered software pipeline: wave 0 issues TDM loads for tile t+1, then waits
// TENSORcnt<=2 (TDM completes in order, so the tile-t pair has landed) before the
// workgroup computes tile t. Epilogue: x = acc*scale[n]+shift[n] (+resid) -> act -> C.
__global__ __launch_bounds__(256)
void k_gemm(const bf16_t* __restrict__ A, const bf16_t* __restrict__ Bt,
            float* __restrict__ C, int M, int N, int Kp,
            const float* __restrict__ scale, const float* __restrict__ shift,
            const float* __restrict__ resid, int flags) {
  __shared__ __align__(16) bf16_t As[2][128 * 32];
  __shared__ __align__(16) bf16_t Bs[2][128 * 32];
  const int tid = threadIdx.x;
  const int wid = tid >> 5;
  const int lane = tid & 31;
  const int m0 = blockIdx.y * 128;
  const int n0 = blockIdx.x * 128;
  const int wm = (wid & 3) * 32;   // wave row offset (4 waves in M)
  const int wn = (wid >> 2) * 64;  // wave col offset (2 waves in N)

  v8f acc[2][4];
#pragma unroll
  for (int i = 0; i < 2; ++i)
#pragma unroll
    for (int j = 0; j < 4; ++j)
#pragma unroll
      for (int r = 0; r < 8; ++r) acc[i][j][r] = 0.f;

  const int lr = lane & 15;
  const int lh = (lane >> 4) * 8;
  const int nk = Kp >> 5;  // Kp is always a multiple of 32

  if (wid == 0) {  // prologue: stage tile 0 into buffer 0
    tdm_load_2d((unsigned)(unsigned long long)(const void*)&As[0][0],
                A + (size_t)m0 * Kp, M - m0, Kp);
    tdm_load_2d((unsigned)(unsigned long long)(const void*)&Bs[0][0],
                Bt + (size_t)n0 * Kp, N - n0, Kp);
  }

  int cur = 0;
  for (int t = 0; t < nk; ++t) {
    if (wid == 0) {
      if (t + 1 < nk) {  // prefetch tile t+1 into the other buffer
        int k0 = (t + 1) << 5;
        tdm_load_2d((unsigned)(unsigned long long)(const void*)&As[cur ^ 1][0],
                    A + (size_t)m0 * Kp + k0, M - m0, Kp);
        tdm_load_2d((unsigned)(unsigned long long)(const void*)&Bs[cur ^ 1][0],
                    Bt + (size_t)n0 * Kp + k0, N - n0, Kp);
        __builtin_amdgcn_s_wait_tensorcnt(2);  // tile t's pair has completed (in-order)
      } else {
        __builtin_amdgcn_s_wait_tensorcnt(0);
      }
    }
    __syncthreads();  // publish buffer `cur` to all waves

    v16bf af[2], bf[4];
#pragma unroll
    for (int q = 0; q < 2; ++q) af[q] = ld_frag(&As[cur][(wm + q * 16 + lr) * 32 + lh]);
#pragma unroll
    for (int q = 0; q < 4; ++q) bf[q] = ld_frag(&Bs[cur][(wn + q * 16 + lr) * 32 + lh]);
#pragma unroll
    for (int i = 0; i < 2; ++i)
#pragma unroll
      for (int j = 0; j < 4; ++j)
        acc[i][j] = __builtin_amdgcn_wmma_f32_16x16x32_bf16(
            false, af[i], false, bf[j], (short)0, acc[i][j], false, false);
    __syncthreads();  // all reads of `cur` done before wave 0 overwrites it next iter
    cur ^= 1;
  }

  const int mrow_add = (lane >> 4) * 8;  // C layout: lanes>=16 carry rows M=8..15
#pragma unroll
  for (int i = 0; i < 2; ++i) {
#pragma unroll
    for (int j = 0; j < 4; ++j) {
      int n = n0 + wn + j * 16 + lr;
      if (n >= N) continue;
      float sc = (flags & EF_SCALE) ? scale[n] : 1.f;
      float sh = (flags & EF_SHIFT) ? shift[n] : 0.f;
#pragma unroll
      for (int r = 0; r < 8; ++r) {
        int m = m0 + wm + i * 16 + mrow_add + r;
        if (m >= M) continue;
        float x = acc[i][j][r] * sc + sh;
        if (flags & EF_RES) x += resid[(size_t)m * N + n];
        if (flags & EF_GELU) x = gelu_exact(x);
        if (flags & EF_SIG) x = 1.f / (1.f + __expf(-x));
        C[(size_t)m * N + n] = x;
      }
    }
  }
}

// ---------------- data movement / conversion kernels ----------------
__global__ void k_cvt2d(const float* in, bf16_t* out, int Kc, int Kp, size_t n) {
  size_t i = (size_t)blockIdx.x * 256 + threadIdx.x;
  if (i >= n) return;
  int k = (int)(i % Kp);
  size_t r = i / Kp;
  out[i] = (k < Kc) ? (bf16_t)in[r * Kc + k] : (bf16_t)0.f;
}

__global__ void k_cvt_tr(const float* in, bf16_t* out, int K, int N, int Kp, size_t n) {
  size_t i = (size_t)blockIdx.x * 256 + threadIdx.x;
  if (i >= n) return;
  int k = (int)(i % Kp);
  size_t nn = i / Kp;
  out[i] = (k < K) ? (bf16_t)in[(size_t)k * N + nn] : (bf16_t)0.f;
}

// deconv weights: w(ci,co,4,4) -> Bt[co][ci*16 + r*4 + c] = w[ci][co][3-r][3-c]
__global__ void k_wflip(const float* w, bf16_t* out, int Ci, int Co, int Kc, int Kp, size_t n) {
  size_t i = (size_t)blockIdx.x * 256 + threadIdx.x;
  if (i >= n) return;
  int q = (int)(i % Kp);
  int co = (int)(i / Kp);
  bf16_t v = (bf16_t)0.f;
  if (q < Kc) {
    int ci = q >> 4, rr = (q >> 2) & 3, cc = q & 3;
    v = (bf16_t)w[(((size_t)ci * Co + co) * 4 + (3 - rr)) * 4 + (3 - cc)];
  }
  out[i] = v;
}

__global__ void k_im2col(const float* in, bf16_t* out, int B, int H, int W, int Ci,
                         int KH, int KW, int strd, int pad, int OH, int OW,
                         int Kc, int Kp, size_t n) {
  size_t i = (size_t)blockIdx.x * 256 + threadIdx.x;
  if (i >= n) return;
  int k = (int)(i % Kp);
  size_t row = i / Kp;
  if (k >= Kc) { out[i] = (bf16_t)0.f; return; }
  int ox = (int)(row % OW);
  size_t t = row / OW;
  int oy = (int)(t % OH);
  int b = (int)(t / OH);
  int kc = k % KW, kr = (k / KW) % KH, ci = k / (KW * KH);
  int y = oy * strd - pad + kr, x = ox * strd - pad + kc;
  float v = 0.f;
  if (y >= 0 && y < H && x >= 0 && x < W)
    v = in[(((size_t)b * H + y) * W + x) * Ci + ci];
  out[i] = (bf16_t)v;
}

__global__ void k_dilate2(const float* in, float* out, int B, int H, int W, int Ci,
                          int H2, int W2, size_t n) {
  size_t i = (size_t)blockIdx.x * 256 + threadIdx.x;
  if (i >= n) return;
  int c = (int)(i % Ci);
  size_t t = i / Ci;
  int x = (int)(t % W2);
  t /= W2;
  int y = (int)(t % H2);
  int b = (int)(t / H2);
  float v = 0.f;
  if (!(y & 1) && !(x & 1)) v = in[(((size_t)b * H + (y >> 1)) * W + (x >> 1)) * Ci + c];
  out[i] = v;
}

__global__ void k_fold_bn(const float* g, const float* b, const float* m, const float* v,
                          const float* cb, float* scale, float* shift, int C) {
  int c = blockIdx.x * blockDim.x + threadIdx.x;
  if (c >= C) return;
  float inv = g[c] * rsqrtf(v[c] + 1e-5f);
  scale[c] = inv;
  shift[c] = b[c] - m[c] * inv + (cb ? cb[c] * inv : 0.f);
}

__global__ void k_copyf(const float* a, float* b, int n) {
  int i = blockIdx.x * blockDim.x + threadIdx.x;
  if (i < n) b[i] = a[i];
}

__global__ void k_zerof(float* p, size_t n) {
  size_t i = (size_t)blockIdx.x * 256 + threadIdx.x;
  if (i < n) p[i] = 0.f;
}

// ---------------- VQ ----------------
__global__ void k_enorm(const float* emb, float* en, int D, size_t n) {
  size_t v = (size_t)blockIdx.x * 256 + threadIdx.x;
  if (v >= n) return;
  float s = 0.f;
  const float* r = emb + v * D;
  for (int d = 0; d < D; ++d) s += r[d] * r[d];
  en[v] = s;
}

__global__ void k_argmin(const float* S, const float* en, int* idx, int M, int V, size_t n) {
  size_t r = (size_t)blockIdx.x * 256 + threadIdx.x;
  if (r >= n) return;
  const float* row = S + r * V;
  float best = 3.4e38f;
  int bi = 0;
  for (int v = 0; v < V; ++v) {
    float d = en[v] - 2.f * row[v];
    if (d < best) { best = d; bi = v; }
  }
  idx[r] = bi;
}

__global__ void k_gather(const int* idx, const float* emb, float* q, int D, size_t n) {
  size_t i = (size_t)blockIdx.x * 256 + threadIdx.x;
  if (i >= n) return;
  size_t r = i / D;
  int d = (int)(i % D);
  q[i] = emb[(size_t)idx[r] * D + d];
}

__global__ void k_mse(const float* q, const float* f, float* sum, size_t n) {
  size_t i = (size_t)blockIdx.x * 256 + threadIdx.x;
  float d = 0.f;
  if (i < n) d = q[i] - f[i];
  float val = d * d;
  for (int o = 16; o > 0; o >>= 1) val += __shfl_down(val, o);
  if ((threadIdx.x & 31) == 0 && val != 0.f) atomicAdd(sum, val);
}

__global__ void k_setqloss(const float* qsum, float* out) {
  out[0] = 0.25f * qsum[0] / 262144.f;  // CC * (sum_h+sum_v)/(B*256*ED)
}

// ---------------- color space ----------------
__global__ void k_rgb2hsv(const float* img, float* hp, float* sp, float* vp, size_t n) {
  size_t p = (size_t)blockIdx.x * 256 + threadIdx.x;
  if (p >= n) return;
  size_t HW = (size_t)IMGSZ * IMGSZ;
  size_t b_ = p / HW, px = p % HW;
  float r = img[(b_ * 3 + 0) * HW + px];
  float g = img[(b_ * 3 + 1) * HW + px];
  float b = img[(b_ * 3 + 2) * HW + px];
  float maxc = fmaxf(r, fmaxf(g, b)), minc = fminf(r, fminf(g, b));
  float delta = maxc - minc;
  float s = delta / (maxc + 1e-8f);
  float dd = (delta == 0.f) ? 1.f : delta;
  float h;
  if (maxc == r) {
    h = fmodf((g - b) / dd, 6.f);
    if (h < 0.f) h += 6.f;
  } else if (maxc == g) h = (b - r) / dd + 2.f;
  else h = (r - g) / dd + 4.f;
  if (delta == 0.f) h = 0.f;
  h *= 1.0471975511965977f;  // pi/3
  hp[p] = h; sp[p] = s; vp[p] = maxc;
}

__global__ void k_hsv2rgb(const float* h, const float* s, const float* v,
                          float* nchw, float* nhwc, size_t n) {
  size_t p = (size_t)blockIdx.x * 256 + threadIdx.x;
  if (p >= n) return;
  float hh = h[p], ss = s[p], vv = v[p];
  float h6 = hh * 0.954929658551372f;  // 6/(2*pi)
  float fi = floorf(h6);
  float f = h6 - fi;
  int ii = ((int)fi) % 6;
  if (ii < 0) ii += 6;
  float pp = vv * (1.f - ss), qq = vv * (1.f - f * ss), tt = vv * (1.f - (1.f - f) * ss);
  float r, g, b;
  switch (ii) {
    case 0: r = vv; g = tt; b = pp; break;
    case 1: r = qq; g = vv; b = pp; break;
    case 2: r = pp; g = vv; b = tt; break;
    case 3: r = pp; g = qq; b = vv; break;
    case 4: r = tt; g = pp; b = vv; break;
    default: r = vv; g = pp; b = qq; break;
  }
  size_t HW = (size_t)IMGSZ * IMGSZ;
  size_t b_ = p / HW, px = p % HW;
  nchw[(b_ * 3 + 0) * HW + px] = r;
  nchw[(b_ * 3 + 1) * HW + px] = g;
  nchw[(b_ * 3 + 2) * HW + px] = b;
  nhwc[p * 3 + 0] = r; nhwc[p * 3 + 1] = g; nhwc[p * 3 + 2] = b;
}

// ---------------- LUT path ----------------
__global__ void k_embmean(const int* ih, const int* iv, const float* emb, float* tf,
                          int ntok, int D, size_t n) {
  size_t i = (size_t)blockIdx.x * 256 + threadIdx.x;
  if (i >= n) return;
  int b = (int)(i / D), d = (int)(i % D);
  float s = 0.f;
  const int* hp = ih + b * ntok;
  const int* vp = iv + b * ntok;
  for (int t = 0; t < ntok; ++t) {
    s += emb[(size_t)hp[t] * D + d];
    s += emb[(size_t)vp[t] * D + d];
  }
  tf[i] = s / (2.f * ntok);
}

__global__ void k_linsmall(const float* x, const float* w, const float* b, float* y,
                           int di, int dout, int act, size_t n) {
  size_t i = (size_t)blockIdx.x * 256 + threadIdx.x;
  if (i >= n) return;
  int bb = (int)(i / dout), o = (int)(i % dout);
  float s = b[o];
  for (int k = 0; k < di; ++k) s += x[(size_t)bb * di + k] * w[(size_t)k * dout + o];
  if (act) s = gelu_exact(s);
  y[i] = s;
}

__global__ void k_softmax8(float* y, int Bn) {
  int b = blockIdx.x * blockDim.x + threadIdx.x;
  if (b >= Bn) return;
  float* r = y + b * 8;
  float mx = r[0];
  for (int i = 1; i < 8; ++i) mx = fmaxf(mx, r[i]);
  float e[8], s = 0.f;
  for (int i = 0; i < 8; ++i) { e[i] = __expf(r[i] - mx); s += e[i]; }
  for (int i = 0; i < 8; ++i) r[i] = e[i] / s;
}

__global__ void k_fuselut(const float* w, const float* gen, float* fl, size_t n) {
  size_t i = (size_t)blockIdx.x * 256 + threadIdx.x;
  if (i >= n) return;
  int b = (int)(i / (3 * G3CUBE));
  int rem = (int)(i % (3 * G3CUBE));
  int c = rem / G3CUBE, g = rem % G3CUBE;
  float s = 0.f;
  const float* gb = gen + (size_t)b * GENOUT;
  for (int nn = 0; nn < NLUT; ++nn)
    s += w[b * 8 + nn] * gb[(size_t)(nn * 3 + c) * G3CUBE + g];
  fl[i] = s;
}

__global__ void k_applylut(const float* rec, const float* lut, float* out, size_t n) {
  size_t p = (size_t)blockIdx.x * 256 + threadIdx.x;
  if (p >= n) return;
  int b_ = (int)(p / ((size_t)IMGSZ * IMGSZ));
  float co[3];
  for (int c = 0; c < 3; ++c) {
    float v = rec[p * 3 + c] * (GRID3 - 1.f);
    co[c] = fminf(fmaxf(v, 0.f), GRID3 - 1.f);
  }
  int z0 = (int)floorf(co[0]); z0 = z0 > GRID3 - 2 ? GRID3 - 2 : (z0 < 0 ? 0 : z0);
  int y0 = (int)floorf(co[1]); y0 = y0 > GRID3 - 2 ? GRID3 - 2 : (y0 < 0 ? 0 : y0);
  int x0 = (int)floorf(co[2]); x0 = x0 > GRID3 - 2 ? GRID3 - 2 : (x0 < 0 ? 0 : x0);
  float fz = co[0] - z0, fy = co[1] - y0, fx = co[2] - x0;
  int z1 = z0 + 1, y1 = y0 + 1, x1 = x0 + 1;
  const float* lb = lut + (size_t)b_ * 3 * G3CUBE;
  for (int c = 0; c < 3; ++c) {
    const float* lc = lb + (size_t)c * G3CUBE;
    auto at = [&](int zi, int yi, int xi) { return lc[((size_t)zi * GRID3 + yi) * GRID3 + xi]; };
    float v00 = at(z0, y0, x0) * (1 - fx) + at(z0, y0, x1) * fx;
    float v01 = at(z0, y1, x0) * (1 - fx) + at(z0, y1, x1) * fx;
    float v10 = at(z1, y0, x0) * (1 - fx) + at(z1, y0, x1) * fx;
    float v11 = at(z1, y1, x0) * (1 - fx) + at(z1, y1, x1) * fx;
    out[p * 3 + c] = (v00 * (1 - fy) + v01 * fy) * (1 - fz) + (v10 * (1 - fy) + v11 * fy) * fz;
  }
}

__global__ void k_nchw2nhwc(const float* in, float* out, size_t n) {
  size_t i = (size_t)blockIdx.x * 256 + threadIdx.x;
  if (i >= n) return;
  size_t HW = (size_t)IMGSZ * IMGSZ;
  int c = (int)(i % 3);
  size_t p = i / 3;
  size_t b = p / HW, px = p % HW;
  out[i] = in[(b * 3 + c) * HW + px];
}

__global__ void k_concat6(const float* a, const float* b, float* out, size_t n) {
  size_t i = (size_t)blockIdx.x * 256 + threadIdx.x;
  if (i >= n) return;
  int c = (int)(i % 6);
  size_t p = i / 6;
  out[i] = (c < 3) ? a[p * 3 + c] : b[p * 3 + (c - 3)];
}

__global__ void k_addimg(const float* stk, const float* img, float* out, size_t n) {
  size_t i = (size_t)blockIdx.x * 256 + threadIdx.x;
  if (i >= n) return;
  size_t HW = (size_t)IMGSZ * IMGSZ;
  int c = (int)(i % 3);
  size_t p = i / 3;
  size_t b = p / HW, px = p % HW;
  size_t o = (b * 3 + c) * HW + px;
  out[o] = stk[i] + img[o];
}

// ---------------- host-side orchestration ----------------
struct Bump {
  char* base;
  size_t off;
  void* take(size_t bytes) {
    size_t a = (off + 255) & ~(size_t)255;
    off = a + bytes;
    return base + a;
  }
};

struct ConvScratch {
  bf16_t* col;
  bf16_t* wbt;
  float* scale;
  float* shift;
};

#define L1D(kern, N, ...)                                           \
  do {                                                              \
    size_t _n = (size_t)(N);                                        \
    if (_n) {                                                       \
      dim3 _g((unsigned)((_n + 255) / 256));                        \
      kern<<<_g, 256, 0, st>>>(__VA_ARGS__);                        \
    }                                                               \
  } while (0)

static inline int rnd32(int x) { return (x + 31) & ~31; }

static void convL(hipStream_t st, ConvScratch& cs, const float* in, int B, int H, int W,
                  int Ci, const float* w, const float* cb, const float* bg, const float* bb,
                  const float* bm, const float* bv, int Co, int kh, int strd, int pad,
                  float* out, int act, const float* resid, int* pOH) {
  int OH = (H + 2 * pad - kh) / strd + 1;
  int OW = (W + 2 * pad - kh) / strd + 1;
  int Kc = Ci * kh * kh, Kp = rnd32(Kc);
  size_t rows = (size_t)B * OH * OW;
  L1D(k_cvt2d, (size_t)Co * Kp, w, cs.wbt, Kc, Kp, (size_t)Co * Kp);
  int flags = EF_SHIFT | act;
  if (bg) {
    flags |= EF_SCALE;
    k_fold_bn<<<dim3((Co + 255) / 256), 256, 0, st>>>(bg, bb, bm, bv, cb, cs.scale, cs.shift, Co);
  } else {
    k_copyf<<<dim3((Co + 255) / 256), 256, 0, st>>>(cb, cs.shift, Co);
  }
  L1D(k_im2col, rows * Kp, in, cs.col, B, H, W, Ci, kh, kh, strd, pad, OH, OW, Kc, Kp, rows * Kp);
  if (resid) flags |= EF_RES;
  dim3 gg((unsigned)((Co + 127) / 128), (unsigned)((rows + 127) / 128));
  k_gemm<<<gg, 256, 0, st>>>(cs.col, cs.wbt, out, (int)rows, Co, Kp, cs.scale, cs.shift, resid, flags);
  *pOH = OH;
}

static void deconvL(hipStream_t st, ConvScratch& cs, float* dil, const float* in, int B,
                    int H, int W, int Ci, const float* w, const float* cb, const float* bg,
                    const float* bb, const float* bm, const float* bv, int Co, float* out,
                    int act) {
  int H2 = 2 * H - 1, W2 = 2 * W - 1;
  L1D(k_dilate2, (size_t)B * H2 * W2 * Ci, in, dil, B, H, W, Ci, H2, W2, (size_t)B * H2 * W2 * Ci);
  int Kc = Ci * 16, Kp = rnd32(Kc);
  L1D(k_wflip, (size_t)Co * Kp, w, cs.wbt, Ci, Co, Kc, Kp, (size_t)Co * Kp);
  int flags = EF_SHIFT | act;
  if (bg) {
    flags |= EF_SCALE;
    k_fold_bn<<<dim3((Co + 255) / 256), 256, 0, st>>>(bg, bb, bm, bv, cb, cs.scale, cs.shift, Co);
  } else {
    k_copyf<<<dim3((Co + 255) / 256), 256, 0, st>>>(cb, cs.shift, Co);
  }
  int OH = 2 * H, OW = 2 * W;
  size_t rows = (size_t)B * OH * OW;
  L1D(k_im2col, rows * Kp, dil, cs.col, B, H2, W2, Ci, 4, 4, 1, 2, OH, OW, Kc, Kp, rows * Kp);
  dim3 gg((unsigned)((Co + 127) / 128), (unsigned)((rows + 127) / 128));
  k_gemm<<<gg, 256, 0, st>>>(cs.col, cs.wbt, out, (int)rows, Co, Kp, cs.scale, cs.shift, nullptr, flags);
}

static void resbL(hipStream_t st, ConvScratch& cs, const float* x, int B, int H, int W,
                  int C, const float* const* IN, int base, float* tmp, float* out) {
  int oh;
  convL(st, cs, x, B, H, W, C, IN[base], IN[base + 1], IN[base + 2], IN[base + 3],
        IN[base + 4], IN[base + 5], C, 3, 1, 1, tmp, EF_GELU, nullptr, &oh);
  convL(st, cs, tmp, B, H, W, C, IN[base + 6], IN[base + 7], IN[base + 8], IN[base + 9],
        IN[base + 10], IN[base + 11], C, 3, 1, 1, out, EF_GELU, x, &oh);
}

extern "C" void kernel_launch(void* const* d_in, const int* in_sizes, int n_in,
                              void* d_out, int out_size, void* d_ws, size_t ws_size,
                              hipStream_t stream) {
  (void)in_sizes; (void)out_size; (void)ws_size;
  if (n_in < 161) return;
  hipStream_t st = stream;
  const float* IN[161];
  for (int i = 0; i < 161; ++i) IN[i] = (const float*)d_in[i];

  const size_t HW = (size_t)IMGSZ * IMGSZ;
  const size_t NPIX = (size_t)BATCH * HW;

  Bump ws{(char*)d_ws, 0};
  float* hpl = (float*)ws.take(NPIX * 4);
  float* spl = (float*)ws.take(NPIX * 4);
  float* vpl = (float*)ws.take(NPIX * 4);
  const size_t ACTN = NPIX * 32;
  float* a0 = (float*)ws.take(ACTN * 4);
  float* a1 = (float*)ws.take(ACTN * 4);
  float* a2 = (float*)ws.take(ACTN * 4);
  float* dil = (float*)ws.take((size_t)BATCH * 255 * 255 * 64 * 4);
  ConvScratch cs;
  cs.col = (bf16_t*)ws.take(NPIX * 1024 * 2);
  cs.wbt = (bf16_t*)ws.take((size_t)256 * 4096 * 2);
  cs.scale = (float*)ws.take(1024 * 4);
  cs.shift = (float*)ws.take(1024 * 4);
  float* fout = (float*)ws.take((size_t)1024 * EDIM * 4);
  bf16_t* Abf = (bf16_t*)ws.take((size_t)1024 * EDIM * 2);
  bf16_t* embbt = (bf16_t*)ws.take((size_t)VCODES * EDIM * 2);
  float* scores = (float*)ws.take((size_t)1024 * VCODES * 4);
  float* enormv = (float*)ws.take(VCODES * 4);
  int* idxh = (int*)ws.take(1024 * 4);
  int* idxv = (int*)ws.take(1024 * 4);
  float* qbuf = (float*)ws.take((size_t)1024 * EDIM * 4);
  float* rech = (float*)ws.take(NPIX * 4);
  float* recv = (float*)ws.take(NPIX * 4);
  float* qsum = (float*)ws.take(256);
  float* recon_nhwc = (float*)ws.take(NPIX * 3 * 4);
  float* img_nhwc = (float*)ws.take(NPIX * 3 * 4);
  float* tf0 = (float*)ws.take(BATCH * HDIM * 4);
  float* tf1 = (float*)ws.take(BATCH * HDIM * 4);
  float* tf2 = (float*)ws.take(BATCH * HDIM * 4);
  float* tf3 = (float*)ws.take(BATCH * HDIM * 4);
  float* wgh = (float*)ws.take(BATCH * 64 * 4);
  float* wsm = (float*)ws.take(BATCH * 8 * 4);
  bf16_t* genwbt = (bf16_t*)ws.take((size_t)GENOUT * HDIM * 2);
  float* genout = (float*)ws.take((size_t)BATCH * GENOUT * 4);
  float* flut = (float*)ws.take((size_t)BATCH * 3 * G3CUBE * 4);
  float* fusedimg = (float*)ws.take(NPIX * 3 * 4);
  float* proc = (float*)ws.take(NPIX * 3 * 4);
  float* cc6 = (float*)ws.take(NPIX * 6 * 4);
  float* stk = (float*)ws.take(NPIX * 3 * 4);

  float* out_recon = (float*)d_out;
  float* out_lut = out_recon + 786432;
  float* out_qloss = out_recon + 2 * 786432;

  // ---- prep ----
  L1D(k_zerof, 1, qsum, (size_t)1);
  L1D(k_rgb2hsv, NPIX, IN[0], hpl, spl, vpl, NPIX);
  L1D(k_cvt2d, (size_t)VCODES * EDIM, IN[115], embbt, EDIM, EDIM, (size_t)VCODES * EDIM);
  L1D(k_enorm, VCODES, IN[115], enormv, EDIM, (size_t)VCODES);

  int oh;
  // ---- two VQ-VAE branches: h (encode->vq->decode) and v ----
  for (int br = 0; br < 2; ++br) {
    const float* plane = br ? vpl : hpl;
    int* idxb = br ? idxv : idxh;
    float* recb = br ? recv : rech;
    // encoder
    convL(st, cs, plane, BATCH, 256, 256, 1, IN[1], IN[2], IN[3], IN[4], IN[5], IN[6],
          64, 4, 2, 1, a0, EF_GELU, nullptr, &oh);                       // 128x128x64
    resbL(st, cs, a0, BATCH, 128, 128, 64, IN, 7, a1, a2);
    convL(st, cs, a2, BATCH, 128, 128, 64, IN[19], IN[20], IN[21], IN[22], IN[23], IN[24],
          128, 4, 2, 1, a0, EF_GELU, nullptr, &oh);                      // 64x64x128
    resbL(st, cs, a0, BATCH, 64, 64, 128, IN, 25, a1, a2);
    convL(st, cs, a2, BATCH, 64, 64, 128, IN[37], IN[38], IN[39], IN[40], IN[41], IN[42],
          256, 4, 2, 1, a0, EF_GELU, nullptr, &oh);                      // 32x32x256
    resbL(st, cs, a0, BATCH, 32, 32, 256, IN, 43, a1, a2);
    convL(st, cs, a2, BATCH, 32, 32, 256, IN[55], IN[56], nullptr, nullptr, nullptr,
          nullptr, EDIM, 4, 2, 1, fout, 0, nullptr, &oh);                // 16x16xED flat
    // VQ: scores = flat @ emb^T via WMMA, then argmin(||e||^2 - 2s)
    L1D(k_cvt2d, (size_t)1024 * EDIM, fout, Abf, EDIM, EDIM, (size_t)1024 * EDIM);
    {
      dim3 gg(8, 8);
      k_gemm<<<gg, 256, 0, st>>>(Abf, embbt, scores, 1024, VCODES, EDIM, nullptr, nullptr,
                                 nullptr, 0);
    }
    L1D(k_argmin, 1024, scores, enormv, idxb, 1024, VCODES, (size_t)1024);
    L1D(k_gather, (size_t)1024 * EDIM, idxb, IN[115], qbuf, EDIM, (size_t)1024 * EDIM);
    L1D(k_mse, (size_t)1024 * EDIM, qbuf, fout, qsum, (size_t)1024 * EDIM);
    // decoder: 1x1 pq conv then 4 deconv stages + resblocks
    convL(st, cs, qbuf, BATCH, 16, 16, EDIM, IN[57], IN[58], nullptr, nullptr, nullptr,
          nullptr, 256, 1, 1, 0, a0, 0, nullptr, &oh);
    deconvL(st, cs, dil, a0, BATCH, 16, 16, 256, IN[59], IN[60], IN[61], IN[62], IN[63],
            IN[64], 256, a1, EF_GELU);                                   // 32
    resbL(st, cs, a1, BATCH, 32, 32, 256, IN, 65, a0, a2);
    deconvL(st, cs, dil, a2, BATCH, 32, 32, 256, IN[77], IN[78], IN[79], IN[80], IN[81],
            IN[82], 128, a0, EF_GELU);                                   // 64
    resbL(st, cs, a0, BATCH, 64, 64, 128, IN, 83, a1, a2);
    deconvL(st, cs, dil, a2, BATCH, 64, 64, 128, IN[95], IN[96], IN[97], IN[98], IN[99],
            IN[100], 64, a0, EF_GELU);                                   // 128
    resbL(st, cs, a0, BATCH, 128, 128, 64, IN, 101, a1, a2);
    deconvL(st, cs, dil, a2, BATCH, 128, 128, 64, IN[113], IN[114], nullptr, nullptr,
            nullptr, nullptr, 1, recb, EF_SIG);                          // 256, sigmoid
  }

  L1D(k_hsv2rgb, NPIX, rech, spl, recv, out_recon, recon_nhwc, NPIX);
  k_setqloss<<<1, 1, 0, st>>>(qsum, out_qloss);

  // ---- LUT generator ----
  L1D(k_embmean, (size_t)BATCH * EDIM, idxh, idxv, IN[116], tf0, 256, EDIM,
      (size_t)BATCH * EDIM);
  L1D(k_linsmall, (size_t)BATCH * HDIM, tf0, IN[117], IN[118], tf1, EDIM, HDIM, 1,
      (size_t)BATCH * HDIM);
  L1D(k_linsmall, (size_t)BATCH * HDIM, tf1, IN[119], IN[120], tf2, HDIM, HDIM, 0,
      (size_t)BATCH * HDIM);
  L1D(k_linsmall, (size_t)BATCH * HDIM, tf2, IN[121], IN[122], tf3, HDIM, HDIM, 1,
      (size_t)BATCH * HDIM);
  // big generator layer: (4,256) x (256,862488) -> WMMA (weight streamed as bf16)
  L1D(k_cvt2d, (size_t)BATCH * HDIM, tf3, Abf, HDIM, HDIM, (size_t)BATCH * HDIM);
  L1D(k_cvt_tr, (size_t)GENOUT * HDIM, IN[123], genwbt, HDIM, GENOUT, HDIM,
      (size_t)GENOUT * HDIM);
  {
    dim3 gg((unsigned)((GENOUT + 127) / 128), 1);
    k_gemm<<<gg, 256, 0, st>>>(Abf, genwbt, genout, BATCH, GENOUT, HDIM, nullptr, IN[124],
                               nullptr, EF_SHIFT);
  }
  L1D(k_linsmall, (size_t)BATCH * 64, tf2, IN[125], IN[126], wgh, HDIM, 64, 1,
      (size_t)BATCH * 64);
  L1D(k_linsmall, (size_t)BATCH * 8, wgh, IN[127], IN[128], wsm, 64, 8, 0,
      (size_t)BATCH * 8);
  k_softmax8<<<1, 32, 0, st>>>(wsm, BATCH);
  L1D(k_fuselut, (size_t)BATCH * 3 * G3CUBE, wsm, genout, flut, (size_t)BATCH * 3 * G3CUBE);
  L1D(k_applylut, NPIX, recon_nhwc, flut, fusedimg, NPIX);
  L1D(k_nchw2nhwc, NPIX * 3, IN[0], img_nhwc, NPIX * 3);
  // rp small stack
  convL(st, cs, img_nhwc, BATCH, IMGSZ, IMGSZ, 3, IN[129], IN[130], nullptr, nullptr,
        nullptr, nullptr, 32, 3, 1, 1, a0, EF_GELU, nullptr, &oh);
  resbL(st, cs, a0, BATCH, IMGSZ, IMGSZ, 32, IN, 131, a1, a2);
  convL(st, cs, a2, BATCH, IMGSZ, IMGSZ, 32, IN[143], IN[144], nullptr, nullptr, nullptr,
        nullptr, 3, 3, 1, 1, proc, 0, nullptr, &oh);
  // fuse small stack
  L1D(k_concat6, NPIX * 6, fusedimg, proc, cc6, NPIX * 6);
  convL(st, cs, cc6, BATCH, IMGSZ, IMGSZ, 6, IN[145], IN[146], nullptr, nullptr, nullptr,
        nullptr, 32, 3, 1, 1, a0, EF_GELU, nullptr, &oh);
  resbL(st, cs, a0, BATCH, IMGSZ, IMGSZ, 32, IN, 147, a1, a2);
  convL(st, cs, a2, BATCH, IMGSZ, IMGSZ, 32, IN[159], IN[160], nullptr, nullptr, nullptr,
        nullptr, 3, 3, 1, 1, stk, 0, nullptr, &oh);
  L1D(k_addimg, NPIX * 3, stk, IN[0], out_lut, NPIX * 3);
}
